// MultiHeadAttention_68822555951319
// MI455X (gfx1250) — matrix operand order repaired and checked by reading, not compiled, not run on hardware
//
#include <hip/hip_runtime.h>
#include <hip/hip_bf16.h>

// ---------------------------------------------------------------------------
// MI455X (gfx1250) wave32 WMMA implementation of the heads-axis attention.
// ---------------------------------------------------------------------------

typedef __attribute__((ext_vector_type(16))) _Float16 v16h;
typedef __attribute__((ext_vector_type(8)))  float    v8f;

#define WMMA_F16(a, b, c) \
  __builtin_amdgcn_wmma_f32_16x16x32_f16(false, (a), false, (b), (short)0, (c), false, false)

// 16-bit A-matrix (16x32) K-base for VGPR-pair i in lane-half `half`
// (ISA 7.12.2: lanes0-15 V0..3 -> K0..7, V4..7 -> K16..23; lanes16-31 +8)
__device__ __forceinline__ int kbA(int i, int half) {
  return (i < 4 ? 2 * i : 16 + 2 * (i - 4)) + 8 * half;
}
// 16-bit B-matrix (32x16) K-base: same K order as A (V0..3 -> K0..7, V4..7 -> K16..23; lanes16-31 +8)
__device__ __forceinline__ int kbB(int i, int half) {
  return (i < 4 ? 2 * i : 16 + 2 * (i - 4)) + 8 * half;
}

// ---------------------------------------------------------------------------
// Kernel 0: Wo f32 -> f16 (row-major, Wo[n][k]; B-frag element (K,N)=Wo[N][K])
// ---------------------------------------------------------------------------
__global__ __launch_bounds__(256) void cvt_wo(const float* __restrict__ w,
                                              _Float16* __restrict__ o, int n) {
  int i = blockIdx.x * 256 + threadIdx.x;
  if (i < n) o[i] = (_Float16)w[i];
}

// ---------------------------------------------------------------------------
// Kernel 1: per-position heads-attention.  One wave per (b,s); 4 waves/block.
// Writes ctx as f16 into X16 in the reference's (buggy) reshape layout:
//   X16[b, h*256 + s/16, (s%16)*64 + d]
// ---------------------------------------------------------------------------
__global__ __launch_bounds__(128) void mha_heads(
    const float* __restrict__ Q, const float* __restrict__ Kk,
    const float* __restrict__ V,
    const float* __restrict__ Wq, const float* __restrict__ bq,
    const float* __restrict__ Wk, const float* __restrict__ bk,
    const float* __restrict__ Wv, const float* __restrict__ bv,
    _Float16* __restrict__ X16) {
  __shared__ alignas(16) _Float16 s_w[3 * 4096];      // Wq,Wk,Wv f16 (24 KB)
  __shared__ alignas(16) _Float16 s_qkv[4][3 * 1024]; // per-wave Qp,Kp,Vp
  __shared__ alignas(16) _Float16 s_attn[4][256];     // per-wave 16x16 attn
  __shared__ alignas(16) _Float16 s_ctx[4][1024];     // per-wave 16x64 ctx

  const int tid = threadIdx.x;
  for (int idx = tid; idx < 4096; idx += 128) {       // stage weights once
    s_w[idx]        = (_Float16)Wq[idx];
    s_w[4096 + idx] = (_Float16)Wk[idx];
    s_w[8192 + idx] = (_Float16)Wv[idx];
  }
  __syncthreads();

  const int wid  = tid >> 5;
  const int lane = tid & 31;
  const int M    = lane & 15;   // row for A-frags / col N for B,C,D frags
  const int half = lane >> 4;

  const int p = blockIdx.x * 4 + wid;   // global position index
  const int b = p >> 12;                // 4096 positions per batch
  const int s = p & 4095;
  const size_t rowoff = ((size_t)b * 4096 + s) * 1024;

  _Float16* qp   = s_qkv[wid];
  _Float16* kp   = s_qkv[wid] + 1024;
  _Float16* vp   = s_qkv[wid] + 2048;
  _Float16* attn = s_attn[wid];
  _Float16* ctx  = s_ctx[wid];

  // -------- Q/K/V projections: dst[h][j] = sum_i x[h][i]*W[j][i] + bias[j]
  for (int m = 0; m < 3; ++m) {
    const float* xrow = (m == 0 ? Q : (m == 1 ? Kk : V)) + rowoff;
    const _Float16* w = s_w + m * 4096;               // W[j][i], row-major
    const float* bias = (m == 0 ? bq : (m == 1 ? bk : bv));
    _Float16* dst = s_qkv[wid] + m * 1024;

    v16h a0, a1;                                      // A = x, 16 heads x 64
#pragma unroll
    for (int i = 0; i < 8; ++i) {
      int kb = kbA(i, half);
      a0[2 * i]     = (_Float16)xrow[M * 64 + kb];
      a0[2 * i + 1] = (_Float16)xrow[M * 64 + kb + 1];
      a1[2 * i]     = (_Float16)xrow[M * 64 + 32 + kb];
      a1[2 * i + 1] = (_Float16)xrow[M * 64 + 32 + kb + 1];
    }
#pragma unroll
    for (int t = 0; t < 4; ++t) {                     // 4 N-tiles of 16 cols
      v8f acc;
#pragma unroll
      for (int i = 0; i < 8; ++i) acc[i] = 0.f;
      v16h bf;
#pragma unroll
      for (int i = 0; i < 8; ++i) {                   // B[K=i][N=j] = W[j][i]
        int kb = kbB(i, half);
        bf[2 * i]     = w[(t * 16 + M) * 64 + kb];
        bf[2 * i + 1] = w[(t * 16 + M) * 64 + kb + 1];
      }
      acc = WMMA_F16(a0, bf, acc);
#pragma unroll
      for (int i = 0; i < 8; ++i) {
        int kb = 32 + kbB(i, half);
        bf[2 * i]     = w[(t * 16 + M) * 64 + kb];
        bf[2 * i + 1] = w[(t * 16 + M) * 64 + kb + 1];
      }
      acc = WMMA_F16(a1, bf, acc);
      float bj = bias[t * 16 + M];
#pragma unroll
      for (int i = 0; i < 8; ++i)                     // D: row = half*8+i
        dst[(half * 8 + i) * 64 + t * 16 + M] = (_Float16)(acc[i] + bj);
    }
  }

  // -------- logits[hq][hk] = (1/8) * sum_d Qp[hq][d]*Kp[hk][d]
  {
    v16h a0, a1, bf;
#pragma unroll
    for (int i = 0; i < 8; ++i) {
      int kb = kbA(i, half);
      a0[2 * i]     = qp[M * 64 + kb];
      a0[2 * i + 1] = qp[M * 64 + kb + 1];
      a1[2 * i]     = qp[M * 64 + 32 + kb];
      a1[2 * i + 1] = qp[M * 64 + 32 + kb + 1];
    }
    v8f lg;
#pragma unroll
    for (int i = 0; i < 8; ++i) lg[i] = 0.f;
#pragma unroll
    for (int i = 0; i < 8; ++i) {                     // B[K=d][N=hk]=Kp[hk][d]
      int kb = kbB(i, half);
      bf[2 * i]     = kp[M * 64 + kb];
      bf[2 * i + 1] = kp[M * 64 + kb + 1];
    }
    lg = WMMA_F16(a0, bf, lg);
#pragma unroll
    for (int i = 0; i < 8; ++i) {
      int kb = 32 + kbB(i, half);
      bf[2 * i]     = kp[M * 64 + kb];
      bf[2 * i + 1] = kp[M * 64 + kb + 1];
    }
    lg = WMMA_F16(a1, bf, lg);

    // softmax over hk: each row (half*8+i) lives across the 16 lanes of a half
#pragma unroll
    for (int i = 0; i < 8; ++i) {
      float v = lg[i] * 0.125f;                       // 1/sqrt(64)
      float mx = v;
      for (int d = 1; d < 16; d <<= 1)
        mx = fmaxf(mx, __shfl_xor(mx, d, 32));
      float e = __expf(v - mx);
      float sm = e;
      for (int d = 1; d < 16; d <<= 1)
        sm += __shfl_xor(sm, d, 32);
      attn[(half * 8 + i) * 16 + M] = (_Float16)(e / sm);
    }
  }

  // -------- ctx[hq][d] = sum_hv attn[hq][hv] * Vp[hv][d]  (K=16, pad to 32)
  {
    v16h aa;
#pragma unroll
    for (int i = 0; i < 8; ++i) {
      int kb = kbA(i, half);
      if (kb < 16) {
        aa[2 * i]     = attn[M * 16 + kb];
        aa[2 * i + 1] = attn[M * 16 + kb + 1];
      } else {
        aa[2 * i]     = (_Float16)0.f;
        aa[2 * i + 1] = (_Float16)0.f;
      }
    }
#pragma unroll
    for (int t = 0; t < 4; ++t) {
      v16h bb;
#pragma unroll
      for (int i = 0; i < 8; ++i) {
        int k = kbB(i, half);                          // K>=16 (elements 8..15, both halves) -> 0
        if (k < 16) {
          bb[2 * i]     = vp[k * 64 + t * 16 + M];
          bb[2 * i + 1] = vp[(k + 1) * 64 + t * 16 + M];
        } else {
          bb[2 * i]     = (_Float16)0.f;
          bb[2 * i + 1] = (_Float16)0.f;
        }
      }
      v8f c0;
#pragma unroll
      for (int i = 0; i < 8; ++i) c0[i] = 0.f;
      c0 = WMMA_F16(aa, bb, c0);
#pragma unroll
      for (int i = 0; i < 8; ++i)
        ctx[(half * 8 + i) * 64 + t * 16 + M] = (_Float16)c0[i];
    }
  }

  // -------- store ctx in the reshaped layout; 64B contiguous per lane
  {
    const int h = lane >> 1, seg = lane & 1;
    const _Float16* src = ctx + h * 64 + seg * 32;
    const size_t row = (size_t)b * 4096 + (size_t)(h * 256 + (s >> 4));
    _Float16* dst = X16 + row * 1024 + (s & 15) * 64 + seg * 32;
#pragma unroll
    for (int j = 0; j < 4; ++j)
      ((uint4*)dst)[j] = ((const uint4*)src)[j];       // 4 x b128
  }
}

// ---------------------------------------------------------------------------
// Kernel 2: out[m][n] = sum_k X16[m][k]*Wo[n][k] + bo[n]
// 256 threads = 8 waves; block tile 128(M) x 128(N), K-step 32.
// LDS double-buffered: next tile's global loads issue before the WMMA burst.
// ---------------------------------------------------------------------------
__global__ __launch_bounds__(256) void gemm_wo(
    const _Float16* __restrict__ X, const _Float16* __restrict__ W,
    const float* __restrict__ bo, float* __restrict__ out) {
  __shared__ alignas(16) _Float16 As[2][128 * 32];  // As[m][k]
  __shared__ alignas(16) _Float16 Bs[2][128 * 32];  // Bs[n][k] (= Wo[n][k])

  const int tid  = threadIdx.x;
  const int lane = tid & 31;
  const int wid  = tid >> 5;
  const int N    = lane & 15;
  const int half = lane >> 4;
  const int n0   = blockIdx.x * 128;
  const int m0   = blockIdx.y * 128;

  // per-thread tile-load coordinates: 2 x uint4 (16 f16) each for A and B
  const int r0 = tid >> 2;             // rows 0..63
  const int r1 = r0 + 64;              // rows 64..127
  const int c0 = (tid & 3) * 8;
  const int l0 = r0 * 32 + c0;         // LDS f16 index
  const int l1 = r1 * 32 + c0;
  const size_t xa0 = (size_t)(m0 + r0) * 1024 + c0;
  const size_t xa1 = (size_t)(m0 + r1) * 1024 + c0;
  const size_t wa0 = (size_t)(n0 + r0) * 1024 + c0;
  const size_t wa1 = (size_t)(n0 + r1) * 1024 + c0;

  v8f acc[8];
#pragma unroll
  for (int t = 0; t < 8; ++t)
#pragma unroll
    for (int i = 0; i < 8; ++i) acc[t][i] = 0.f;

  // preload K-tile 0 into buffer 0
  *(uint4*)&As[0][l0] = *(const uint4*)&X[xa0];
  *(uint4*)&As[0][l1] = *(const uint4*)&X[xa1];
  *(uint4*)&Bs[0][l0] = *(const uint4*)&W[wa0];
  *(uint4*)&Bs[0][l1] = *(const uint4*)&W[wa1];
  __syncthreads();

  for (int k0 = 0; k0 < 1024; k0 += 32) {
    const int cur = (k0 >> 5) & 1;
    const int nxt = cur ^ 1;
    const bool more = (k0 + 32) < 1024;

    // issue next tile's global loads first (latency hidden by WMMAs below)
    uint4 ra0, ra1, rb0, rb1;
    if (more) {
      ra0 = *(const uint4*)&X[xa0 + k0 + 32];
      ra1 = *(const uint4*)&X[xa1 + k0 + 32];
      rb0 = *(const uint4*)&W[wa0 + k0 + 32];
      rb1 = *(const uint4*)&W[wa1 + k0 + 32];
      if (k0 + 64 < 1024) {                 // prime GL2 one tile further out
        __builtin_prefetch(&X[xa0 + k0 + 64], 0, 1);
        __builtin_prefetch(&W[wa0 + k0 + 64], 0, 1);
      }
    }

    // compute on current buffer
    v16h a;
#pragma unroll
    for (int i = 0; i < 8; ++i) {
      int kb = kbA(i, half);
      a[2 * i]     = As[cur][(wid * 16 + N) * 32 + kb];   // A row = lane&15
      a[2 * i + 1] = As[cur][(wid * 16 + N) * 32 + kb + 1];
    }
#pragma unroll
    for (int t = 0; t < 8; ++t) {
      v16h bf;
#pragma unroll
      for (int i = 0; i < 8; ++i) {
        int kb = kbB(i, half);
        bf[2 * i]     = Bs[cur][(t * 16 + N) * 32 + kb];
        bf[2 * i + 1] = Bs[cur][(t * 16 + N) * 32 + kb + 1];
      }
      acc[t] = WMMA_F16(a, bf, acc[t]);
    }

    // stage next tile into the other buffer, then one barrier per iteration
    if (more) {
      *(uint4*)&As[nxt][l0] = ra0;
      *(uint4*)&As[nxt][l1] = ra1;
      *(uint4*)&Bs[nxt][l0] = rb0;
      *(uint4*)&Bs[nxt][l1] = rb1;
    }
    __syncthreads();
  }

#pragma unroll
  for (int t = 0; t < 8; ++t) {
    float bv = bo[n0 + t * 16 + N];
#pragma unroll
    for (int i = 0; i < 8; ++i)
      out[(size_t)(m0 + wid * 16 + half * 8 + i) * 1024 + n0 + t * 16 + N] =
          acc[t][i] + bv;
  }
}

// ---------------------------------------------------------------------------
extern "C" void kernel_launch(void* const* d_in, const int* in_sizes, int n_in,
                              void* d_out, int out_size, void* d_ws,
                              size_t ws_size, hipStream_t stream) {
  const float* Q  = (const float*)d_in[0];
  const float* K  = (const float*)d_in[1];
  const float* V  = (const float*)d_in[2];
  const float* Wq = (const float*)d_in[3];
  const float* bq = (const float*)d_in[4];
  const float* Wk = (const float*)d_in[5];
  const float* bk = (const float*)d_in[6];
  const float* Wv = (const float*)d_in[7];
  const float* bv = (const float*)d_in[8];
  const float* Wo = (const float*)d_in[9];
  const float* bo = (const float*)d_in[10];

  _Float16* X16  = (_Float16*)d_ws;                               // 64 MB
  _Float16* Wo16 = (_Float16*)((char*)d_ws + (((size_t)64) << 20)); // 2 MB

  cvt_wo<<<4096, 256, 0, stream>>>(Wo, Wo16, 1024 * 1024);
  mha_heads<<<8192, 128, 0, stream>>>(Q, K, V, Wq, bq, Wk, bk, Wv, bv, X16);
  gemm_wo<<<dim3(8, 256), 256, 0, stream>>>(X16, Wo16, bo, (float*)d_out);
}
